// SpatialMambaBlock_38534446579789
// MI455X (gfx1250) — matrix operand order repaired and checked by reading, not compiled, not run on hardware
//
#include <hip/hip_runtime.h>

// SpatialMambaBlock fused kernel for gfx1250 (MI455X).
// One block per (batch, H) row: async-copy row to LDS -> LN -> {expand,
// sigmoid-gate} WMMA GEMMs -> in-register SSM scan along W -> gated output
// WMMA GEMM. All intermediates live in LDS (320 KB/WGP); HBM traffic is just
// input row + output row (~101 MB total => ~4.3us floor at 23.3 TB/s).

typedef __attribute__((ext_vector_type(16))) __bf16 bf16x16;
typedef __attribute__((ext_vector_type(8)))  __bf16 bf16x8;
typedef __attribute__((ext_vector_type(8)))  float  f32x8;
typedef __attribute__((ext_vector_type(4)))  float  f32x4;

namespace {
constexpr int kCH = 96;      // channels
constexpr int kE  = 192;     // expanded channels
constexpr int kS  = 16;      // ssm state dim
constexpr int kW  = 128;     // scan length (W dimension)
constexpr int kThreads = 256;
constexpr float kEps = 1e-3f;

// ---- dynamic-LDS byte offsets (all 64B aligned) ----
constexpr int offRaw   = 0;        // f32 [128*96] raw input row; UNION with bf16 gate[128*192]
constexpr int offXln   = 49152;    // bf16[128*96]   normalized input
constexpr int offWexp  = 73728;    // bf16[192*96]   W_expand^T  ([e][c])
constexpr int offWgate = 110592;   // bf16[192*96]   W_gate^T    ([e][c])
constexpr int offWout  = 147456;   // bf16[96*192]   W_out^T     ([c][e])
constexpr int offXproj = 184320;   // bf16[128*192]  x_proj, reused in-place as gated
constexpr int offA     = 233472;   // f32 [192*16]
constexpr int offB     = 245760;   // f32 [192*16]
constexpr int offC     = 258048;   // f32 [192*16]
constexpr int offD     = 270336;   // f32 [192]
constexpr int offGam   = 271104;   // f32 [96]
constexpr int offBet   = 271488;   // f32 [96]
constexpr int kLdsBytes = 271872;  // <= 320 KB WGP LDS
} // namespace

// Async global->LDS copy (CDNA5 ASYNCcnt path, bypasses VGPR staging).
// A generic pointer into LDS carries the wave-relative LDS byte offset in its
// low 32 bits (flat LDS aperture truncation), which is the VDST operand.
__device__ __forceinline__ void async_copy_b128(void* lds_dst, const void* gsrc) {
  unsigned ldsOff = (unsigned)(size_t)lds_dst;
  asm volatile("global_load_async_to_lds_b128 %0, %1, off"
               :: "v"(ldsOff), "v"(gsrc) : "memory");
}
__device__ __forceinline__ void wait_asynccnt0() {
  asm volatile("s_wait_asynccnt 0x0" ::: "memory");
}

// A-fragment (16x32 bf16, M rows striped in-lane): lane L row = m0+(L&15),
// K = k0 + (L[4]?16:0) + {0..15}, contiguous (matches loadB).
__device__ __forceinline__ bf16x16 loadA(const __bf16* base, int ldk, int m0, int k0, int lane) {
  const __bf16* p = base + (m0 + (lane & 15)) * ldk + k0 + ((lane & 16) ? 16 : 0);
  bf16x8 lo = *(const bf16x8*)p;
  bf16x8 hi = *(const bf16x8*)(p + 8);
  return __builtin_shufflevector(lo, hi, 0,1,2,3,4,5,6,7,8,9,10,11,12,13,14,15);
}

// B-fragment (32x16 bf16): lane L col = n0+(L&15), K = k0 + (L[4]?16:0) + {0..15},
// contiguous because the weight is staged transposed ([n][k], stride ldk).
__device__ __forceinline__ bf16x16 loadB(const __bf16* baseT, int ldk, int n0, int k0, int lane) {
  const __bf16* p = baseT + (n0 + (lane & 15)) * ldk + k0 + ((lane & 16) ? 16 : 0);
  return *(const bf16x16*)p;
}

__global__ __launch_bounds__(kThreads)
void spatial_mamba_kernel(const float* __restrict__ inp,
                          const float* __restrict__ gamma,
                          const float* __restrict__ beta,
                          const float* __restrict__ Wexp,
                          const float* __restrict__ Wgat,
                          const float* __restrict__ Amat,
                          const float* __restrict__ Bmat,
                          const float* __restrict__ Cmat,
                          const float* __restrict__ Dvec,
                          const float* __restrict__ Wout,
                          float* __restrict__ out)
{
  extern __shared__ char smem[];
  float*  raw   = (float*) (smem + offRaw);
  __bf16* gateB = (__bf16*)(smem + offRaw);     // union: raw dead after LN
  __bf16* xln   = (__bf16*)(smem + offXln);
  __bf16* wexpT = (__bf16*)(smem + offWexp);
  __bf16* wgatT = (__bf16*)(smem + offWgate);
  __bf16* woutT = (__bf16*)(smem + offWout);
  __bf16* xproj = (__bf16*)(smem + offXproj);
  float*  As    = (float*) (smem + offA);
  float*  Bs    = (float*) (smem + offB);
  float*  Cs    = (float*) (smem + offC);
  float*  Ds    = (float*) (smem + offD);
  float*  gs    = (float*) (smem + offGam);
  float*  bs    = (float*) (smem + offBet);

  const int tid  = threadIdx.x;
  const int lane = tid & 31;
  const int wv   = tid >> 5;
  const long rowBase = (long)blockIdx.x * (kW * kCH);

  // ---------------- Phase A: stage input row (async DMA) + weights ----------------
  {
    const f32x4* src = (const f32x4*)(inp + rowBase);
    f32x4* dst = (f32x4*)raw;
    #pragma unroll
    for (int it = 0; it < (kW * kCH / 4) / kThreads; ++it) {
      const int i = it * kThreads + tid;
      async_copy_b128(dst + i, src + i);          // global_load_async_to_lds_b128
    }
  }
  for (int i = tid; i < kCH * kE; i += kThreads) {       // W_expand^T / W_gate^T
    int e = i / kCH, c = i - e * kCH;
    wexpT[i] = (__bf16)Wexp[c * kE + e];
    wgatT[i] = (__bf16)Wgat[c * kE + e];
  }
  for (int i = tid; i < kCH * kE; i += kThreads) {       // W_out^T
    int c = i / kE, e = i - c * kE;
    woutT[i] = (__bf16)Wout[e * kCH + c];
  }
  for (int i = tid; i < kE * kS / 4; i += kThreads) {    // SSM params (f32, layout kept)
    ((f32x4*)As)[i] = ((const f32x4*)Amat)[i];
    ((f32x4*)Bs)[i] = ((const f32x4*)Bmat)[i];
    ((f32x4*)Cs)[i] = ((const f32x4*)Cmat)[i];
  }
  if (tid < kE)  Ds[tid] = Dvec[tid];
  if (tid < kCH) { gs[tid] = gamma[tid]; bs[tid] = beta[tid]; }
  wait_asynccnt0();                               // s_wait_asynccnt 0
  __syncthreads();

  // ---------------- Phase B: LayerNorm (f32) -> bf16 xln ----------------
  if (tid < kW) {
    const f32x4* r4 = (const f32x4*)(raw + tid * kCH);
    float m = 0.f;
    #pragma unroll
    for (int i = 0; i < kCH / 4; ++i) { f32x4 v = r4[i]; m += v[0] + v[1] + v[2] + v[3]; }
    m *= (1.0f / kCH);
    float var = 0.f;
    #pragma unroll
    for (int i = 0; i < kCH / 4; ++i) {
      f32x4 v = r4[i];
      float d0 = v[0] - m, d1 = v[1] - m, d2 = v[2] - m, d3 = v[3] - m;
      var += d0 * d0 + d1 * d1 + d2 * d2 + d3 * d3;
    }
    var *= (1.0f / kCH);
    const float inv = rsqrtf(var + kEps);
    #pragma unroll
    for (int c = 0; c < kCH; ++c) {
      float v = (raw[tid * kCH + c] - m) * inv * gs[c] + bs[c];
      xln[tid * kCH + c] = (__bf16)v;
    }
  }
  __syncthreads();

  // ------- Phase C: x_proj = xln @ Wexp (WMMA). 8x12 tiles, 12 per wave. -------
  for (int t = wv; t < 96; t += 8) {
    const int m0 = (t / 12) * 16;
    const int n0 = (t % 12) * 16;
    f32x8 acc = {};
    #pragma unroll
    for (int k0 = 0; k0 < kCH; k0 += 32) {
      bf16x16 a = loadA(xln, kCH, m0, k0, lane);
      bf16x16 b = loadB(wexpT, kCH, n0, k0, lane);
      acc = __builtin_amdgcn_wmma_f32_16x16x32_bf16(false, a, false, b, (short)0, acc, false, false);
    }
    const int row = m0 + ((lane & 16) ? 8 : 0);
    const int col = n0 + (lane & 15);
    #pragma unroll
    for (int r = 0; r < 8; ++r) xproj[(row + r) * kE + col] = (__bf16)acc[r];
  }
  // ------- Phase C': gate = sigmoid(xln @ Wgat) (WMMA + v_rcp sigmoid) -------
  for (int t = wv; t < 96; t += 8) {
    const int m0 = (t / 12) * 16;
    const int n0 = (t % 12) * 16;
    f32x8 acc = {};
    #pragma unroll
    for (int k0 = 0; k0 < kCH; k0 += 32) {
      bf16x16 a = loadA(xln, kCH, m0, k0, lane);
      bf16x16 b = loadB(wgatT, kCH, n0, k0, lane);
      acc = __builtin_amdgcn_wmma_f32_16x16x32_bf16(false, a, false, b, (short)0, acc, false, false);
    }
    const int row = m0 + ((lane & 16) ? 8 : 0);
    const int col = n0 + (lane & 15);
    #pragma unroll
    for (int r = 0; r < 8; ++r) {
      const float sg = __builtin_amdgcn_rcpf(1.0f + __expf(-acc[r]));  // v_rcp_f32
      gateB[(row + r) * kE + col] = (__bf16)sg;
    }
  }
  __syncthreads();

  // ---------------- Phase D: SSM scan along W, fused gating ----------------
  // Thread e owns channel e's 16-wide state in registers; writes gated back in place.
  if (tid < kE) {
    float a[kS], bq[kS], cq[kS], st[kS];
    #pragma unroll
    for (int s = 0; s < kS; ++s) {
      a[s]  = As[tid * kS + s];
      bq[s] = Bs[tid * kS + s];
      cq[s] = Cs[tid * kS + s];
      st[s] = 0.f;
    }
    const float d = Ds[tid];
    for (int w = 0; w < kW; ++w) {
      const int idx = w * kE + tid;
      const float u = (float)xproj[idx];
      float y = 0.f;
      #pragma unroll
      for (int s = 0; s < kS; ++s) {
        st[s] = fmaf(st[s], a[s], u * bq[s]);
        y = fmaf(st[s], cq[s], y);
      }
      y = fmaf((float)kS * d, u, y);              // faithful S*u*D term
      const float g = (float)gateB[idx];
      xproj[idx] = (__bf16)(y * g);               // in-place: u[w] is dead now
    }
  }
  __syncthreads();

  // ---------------- Phase E: out = gated @ W_out (WMMA), store f32 ----------------
  for (int t = wv; t < 48; t += 8) {              // 8 M-tiles x 6 N-tiles
    const int m0 = (t / 6) * 16;
    const int n0 = (t % 6) * 16;
    f32x8 acc = {};
    #pragma unroll
    for (int k0 = 0; k0 < kE; k0 += 32) {
      bf16x16 a = loadA(xproj, kE, m0, k0, lane);
      bf16x16 b = loadB(woutT, kE, n0, k0, lane);
      acc = __builtin_amdgcn_wmma_f32_16x16x32_bf16(false, a, false, b, (short)0, acc, false, false);
    }
    const int row = m0 + ((lane & 16) ? 8 : 0);
    const int col = n0 + (lane & 15);
    float* op = out + rowBase;
    #pragma unroll
    for (int r = 0; r < 8; ++r) op[(row + r) * kCH + col] = acc[r];
  }
}

extern "C" void kernel_launch(void* const* d_in, const int* in_sizes, int n_in,
                              void* d_out, int out_size, void* d_ws, size_t ws_size,
                              hipStream_t stream) {
  const float* inp  = (const float*)d_in[0];
  const float* gam  = (const float*)d_in[1];
  const float* bet  = (const float*)d_in[2];
  const float* wexp = (const float*)d_in[3];
  const float* wgat = (const float*)d_in[4];
  const float* amat = (const float*)d_in[5];
  const float* bmat = (const float*)d_in[6];
  const float* cmat = (const float*)d_in[7];
  const float* dvec = (const float*)d_in[8];
  const float* wout = (const float*)d_in[9];
  float* out = (float*)d_out;

  // Opt in to >64KB dynamic LDS (deterministic, capture-safe host call).
  hipFuncSetAttribute((const void*)spatial_mamba_kernel,
                      hipFuncAttributeMaxDynamicSharedMemorySize, kLdsBytes);

  dim3 grid(8 * 128);      // one block per (batch, H) row
  dim3 block(kThreads);    // 8 wave32 waves
  spatial_mamba_kernel<<<grid, block, kLdsBytes, stream>>>(
      inp, gam, bet, wexp, wgat, amat, bmat, cmat, dvec, wout, out);
}
